// MoeLoopBlock_11175504904521
// MI455X (gfx1250) — matrix operand order repaired and checked by reading, not compiled, not run on hardware
//
#include <hip/hip_runtime.h>
#include <hip/hip_bf16.h>
#include <math.h>
#include <stdint.h>

// Problem constants (from reference): B=4, S=2048, D=1024, E=8, F=4096, K=2
#define BDIM 4
#define SDIM 2048
#define DDIM 1024
#define EDIM 8
#define FDIM 4096
#define NTOK (BDIM * SDIM)   // 8192 tokens

typedef __attribute__((ext_vector_type(16))) __bf16 v16bf;
typedef __attribute__((ext_vector_type(8)))  float  v8f;

struct alignas(16) U128 { unsigned int u[4]; };
struct alignas(32) FragBuf { U128 lo, hi; };

// Workspace layout (bytes). Total requirement ~275 MB (HBM is 432 GB).
#define CNT_OFF   0u                                  // E ints
#define TID_OFF   1024u                               // E*NTOK ints   (256 KB)
#define WGT_OFF   (TID_OFF + EDIM * NTOK * 4u)        // E*NTOK floats (256 KB)
#define XB_OFF    (1u << 20)                          // NTOK*D bf16   (16 MB)
#define HB_OFF    (XB_OFF + NTOK * DDIM * 2u + (1u << 20))     // NTOK*F bf16 (64 MB)
#define WT0_OFF   (HB_OFF + NTOK * FDIM * 2u + (1u << 20))     // wi0^T bf16 [E][F][D] (64 MB)
#define WT1_OFF   (WT0_OFF + EDIM * FDIM * DDIM * 2u)          // wi1^T bf16 [E][F][D] (64 MB)
#define WOT_OFF   (WT1_OFF + EDIM * FDIM * DDIM * 2u)          // wo^T  bf16 [E][D][F] (64 MB)

// LDS row strides (elements) — padded to kill bank conflicts on b128 reads
#define LDA 40   // X / H tiles: 128 rows x 32 K
#define LDB 40   // W tiles (transposed): 64 N-rows x 32 K

// LDS offset of a __shared__ pointer: generic pointers to LDS carry the
// wave-relative LDS byte offset in their low 32 bits (ISA aperture rules).
__device__ __forceinline__ unsigned lds_off(const void* p) {
    return (unsigned)(uintptr_t)p;
}
// Async copy 16B global -> LDS (tracked by ASYNCcnt, bypasses VGPRs).
__device__ __forceinline__ void async_cp16(unsigned lds, const __bf16* g) {
    asm volatile("global_load_async_to_lds_b128 %0, %1, off"
                 :: "v"(lds), "v"(g) : "memory");
}
__device__ __forceinline__ void wait_async0() {
    asm volatile("s_wait_asynccnt 0x0" ::: "memory");
}

// ---------------------------------------------------------------------------
__global__ void moe_init(float* __restrict__ out, int n, int* __restrict__ cnt) {
    int i = blockIdx.x * blockDim.x + threadIdx.x;
    if (i < n) out[i] = 0.0f;
    if (i < EDIM) cnt[i] = 0;
}

__global__ void moe_cvt_x(const float* __restrict__ x, __bf16* __restrict__ xb) {
    int i = blockIdx.x * blockDim.x + threadIdx.x;
    xb[i] = (__bf16)x[i];
}

// Transpose+convert: src fp32 [R][C] -> dst bf16 [C][R]. Block (32,8), tile 32x32.
__global__ void moe_cvt_trans(const float* __restrict__ src, __bf16* __restrict__ dst,
                              int R, int C) {
    __shared__ float t[32][33];
    int c0 = blockIdx.x * 32, r0 = blockIdx.y * 32;
    int tx = threadIdx.x, ty = threadIdx.y;
#pragma unroll
    for (int j = ty; j < 32; j += 8)
        t[j][tx] = src[(size_t)(r0 + j) * C + c0 + tx];
    __syncthreads();
#pragma unroll
    for (int j = ty; j < 32; j += 8)
        dst[(size_t)(c0 + j) * R + r0 + tx] = (__bf16)t[tx][j];
}

// One wave32 per token: gate logits, top-2, softmax, scatter to expert lists.
__global__ void moe_gate(const float* __restrict__ x, const float* __restrict__ gw,
                         int* __restrict__ cnt, int* __restrict__ tidl,
                         float* __restrict__ wgt) {
    int lane = threadIdx.x & 31;
    int wave = threadIdx.x >> 5;
    int t = blockIdx.x * 8 + wave;
    const float* xr = x + (size_t)t * DDIM;

    float acc[EDIM];
#pragma unroll
    for (int e = 0; e < EDIM; ++e) acc[e] = 0.0f;

    for (int d = lane; d < DDIM; d += 32) {
        float xv = xr[d];
        const float4* g4 = (const float4*)(gw + (size_t)d * EDIM);
        float4 g0 = g4[0], g1 = g4[1];
        acc[0] += xv * g0.x; acc[1] += xv * g0.y;
        acc[2] += xv * g0.z; acc[3] += xv * g0.w;
        acc[4] += xv * g1.x; acc[5] += xv * g1.y;
        acc[6] += xv * g1.z; acc[7] += xv * g1.w;
    }
#pragma unroll
    for (int e = 0; e < EDIM; ++e)
#pragma unroll
        for (int m = 16; m >= 1; m >>= 1)
            acc[e] += __shfl_xor(acc[e], m, 32);

    if (lane == 0) {
        float b0 = -1e30f, b1 = -1e30f; int i0 = 0, i1 = 0;
#pragma unroll
        for (int e = 0; e < EDIM; ++e) {
            float l = acc[e];
            if (l > b0)      { b1 = b0; i1 = i0; b0 = l; i0 = e; }
            else if (l > b1) { b1 = l;  i1 = e; }
        }
        float e1 = __expf(b1 - b0);
        float inv = 1.0f / (1.0f + e1);
        int s0 = atomicAdd(&cnt[i0], 1);
        tidl[i0 * NTOK + s0] = t; wgt[i0 * NTOK + s0] = inv;
        int s1 = atomicAdd(&cnt[i1], 1);
        tidl[i1 * NTOK + s1] = t; wgt[i1 * NTOK + s1] = e1 * inv;
    }
}

// --------------------------- WMMA fragment helpers -------------------------
// A fragment (16x32 bf16): lane<16 -> row=lane, K in {0..7, 16..23};
//                          lane>=16 -> row=lane-16, K in {8..15, 24..31}.
__device__ __forceinline__ v16bf load_a_frag(const __bf16* base, int lane) {
    int m  = lane & 15;
    int k0 = (lane >> 4) * 8;
    const __bf16* p = base + m * LDA + k0;
    FragBuf fb;
    fb.lo = *(const U128*)p;          // K = k0..k0+7
    fb.hi = *(const U128*)(p + 16);   // K = k0+16..k0+23
    return *(const v16bf*)&fb;
}
// B fragment (32x16 bf16), W staged in LDS as [n][k]:
// lanes 0-15 -> col=lane, K in {0..7, 16..23}; lanes 16-31 -> col=lane-16, K in {8..15, 24..31}.
__device__ __forceinline__ v16bf load_b_frag(const __bf16* base, int lane) {
    int n  = lane & 15;
    int k0 = (lane >> 4) * 8;
    const __bf16* p = base + n * LDB + k0;
    FragBuf fb;
    fb.lo = *(const U128*)p;
    fb.hi = *(const U128*)(p + 16);
    return *(const v16bf*)&fb;
}

// ---------------------------------------------------------------------------
// mlp1: for expert e, h[slot, f] = silu(X@wi0) * (X@wi1), X rows gathered.
// CTA tile: 128 slots x 64 F-cols; K loop over D in steps of 32.
// All operands bf16, staged with double-buffered async global->LDS copies.
__global__ void __launch_bounds__(256)
moe_mlp1(const __bf16* __restrict__ xb, const __bf16* __restrict__ wt0,
         const __bf16* __restrict__ wt1, const int* __restrict__ cnt,
         const int* __restrict__ tidl, __bf16* __restrict__ hb, int e) {
    int ne = cnt[e];
    int mBase = blockIdx.y * 128;
    if (mBase >= ne) return;
    int fBase = blockIdx.x * 64;

    const int* tlist = tidl + e * NTOK;
    const __bf16* W0 = wt0 + (size_t)e * FDIM * DDIM;  // [F][D] bf16
    const __bf16* W1 = wt1 + (size_t)e * FDIM * DDIM;

    __shared__ __align__(32) __bf16 sX[2][128 * LDA];
    __shared__ __align__(32) __bf16 sW0[2][64 * LDB];
    __shared__ __align__(32) __bf16 sW1[2][64 * LDB];

    int tix = threadIdx.x, lane = tix & 31, wave = tix >> 5;
    int waveM = (wave & 3) * 32;   // 0..96
    int waveN = (wave >> 2) * 32;  // 0,32

    // X staging: 2 threads/row, 2x16B per thread. Clamp padded rows to slot 0
    // (their results are never stored).
    int xRow = tix >> 1;
    int xCol = (tix & 1) * 16;
    int slot = mBase + xRow;
    long xtok = (long)tlist[slot < ne ? slot : 0];
    const __bf16* gX = xb + xtok * DDIM + xCol;
    unsigned lX[2] = { lds_off(&sX[0][xRow * LDA + xCol]),
                       lds_off(&sX[1][xRow * LDA + xCol]) };

    // W staging: 64 rows x 32 k -> one 16B chunk per thread per matrix.
    int wn = tix >> 2;
    int wk = (tix & 3) * 8;
    const __bf16* gW0 = W0 + (size_t)(fBase + wn) * DDIM + wk;
    const __bf16* gW1 = W1 + (size_t)(fBase + wn) * DDIM + wk;
    unsigned lW0[2] = { lds_off(&sW0[0][wn * LDB + wk]), lds_off(&sW0[1][wn * LDB + wk]) };
    unsigned lW1[2] = { lds_off(&sW1[0][wn * LDB + wk]), lds_off(&sW1[1][wn * LDB + wk]) };

    auto stage = [&](int k0, int b) {
        async_cp16(lX[b],      gX + k0);
        async_cp16(lX[b] + 16, gX + k0 + 8);
        async_cp16(lW0[b],     gW0 + k0);
        async_cp16(lW1[b],     gW1 + k0);
    };

    v8f c0[2][2], c1[2][2];
#pragma unroll
    for (int a = 0; a < 2; ++a)
#pragma unroll
        for (int b = 0; b < 2; ++b)
#pragma unroll
            for (int i = 0; i < 8; ++i) { c0[a][b][i] = 0.0f; c1[a][b][i] = 0.0f; }

    stage(0, 0);
    const int NK = DDIM / 32;
    for (int idx = 0; idx < NK; ++idx) {
        int b = idx & 1;
        wait_async0();        // my async copies for tile idx have landed in LDS
        __syncthreads();      // everyone's copies visible; buf b^1 reads done
        if (idx + 1 < NK) stage((idx + 1) * 32, b ^ 1);  // prefetch next tile

        const __bf16* bx  = sX[b];
        const __bf16* bw0 = sW0[b];
        const __bf16* bw1 = sW1[b];
        v16bf a0 = load_a_frag(bx + (waveM +  0) * LDA, lane);
        v16bf a1 = load_a_frag(bx + (waveM + 16) * LDA, lane);
#pragma unroll
        for (int ni = 0; ni < 2; ++ni) {
            v16bf b0 = load_b_frag(bw0 + (waveN + ni * 16) * LDB, lane);
            v16bf b1 = load_b_frag(bw1 + (waveN + ni * 16) * LDB, lane);
            c0[0][ni] = __builtin_amdgcn_wmma_f32_16x16x32_bf16(false, a0, false, b0, (short)0, c0[0][ni], false, false);
            c0[1][ni] = __builtin_amdgcn_wmma_f32_16x16x32_bf16(false, a1, false, b0, (short)0, c0[1][ni], false, false);
            c1[0][ni] = __builtin_amdgcn_wmma_f32_16x16x32_bf16(false, a0, false, b1, (short)0, c1[0][ni], false, false);
            c1[1][ni] = __builtin_amdgcn_wmma_f32_16x16x32_bf16(false, a1, false, b1, (short)0, c1[1][ni], false, false);
        }
    }

    // ---- epilogue: h = silu(c0) * c1, store bf16 ----
#pragma unroll
    for (int mi = 0; mi < 2; ++mi)
#pragma unroll
        for (int ni = 0; ni < 2; ++ni)
#pragma unroll
            for (int i = 0; i < 8; ++i) {
                int m = waveM + mi * 16 + ((lane >> 4) * 8 + i);
                int row = mBase + m;
                if (row < ne) {
                    float s = c0[mi][ni][i];
                    float h = (s / (1.0f + __expf(-s))) * c1[mi][ni][i];
                    int f = fBase + waveN + ni * 16 + (lane & 15);
                    hb[(size_t)row * FDIM + f] = (__bf16)h;
                }
            }
}

// ---------------------------------------------------------------------------
// mlp2: y = h @ wo[e]; out[token] += gate_weight * y.  CTA: 128 slots x 64 D.
__global__ void __launch_bounds__(256)
moe_mlp2(const __bf16* __restrict__ hb, const __bf16* __restrict__ wot,
         const int* __restrict__ cnt, const int* __restrict__ tidl,
         const float* __restrict__ wgt, float* __restrict__ out, int e) {
    int ne = cnt[e];
    int mBase = blockIdx.y * 128;
    if (mBase >= ne) return;
    int dBase = blockIdx.x * 64;

    const __bf16* W = wot + (size_t)e * DDIM * FDIM;   // [D][F] bf16

    __shared__ __align__(32) __bf16 sH[2][128 * LDA];
    __shared__ __align__(32) __bf16 sW[2][64 * LDB];

    int tix = threadIdx.x, lane = tix & 31, wave = tix >> 5;
    int waveM = (wave & 3) * 32;
    int waveN = (wave >> 2) * 32;

    int hRow = tix >> 1;
    int hCol = (tix & 1) * 16;
    const __bf16* gH = hb + (size_t)(mBase + hRow) * FDIM + hCol;  // rows>=ne: garbage, never stored
    unsigned lH[2] = { lds_off(&sH[0][hRow * LDA + hCol]),
                       lds_off(&sH[1][hRow * LDA + hCol]) };

    int wn = tix >> 2;
    int wk = (tix & 3) * 8;
    const __bf16* gW = W + (size_t)(dBase + wn) * FDIM + wk;
    unsigned lW[2] = { lds_off(&sW[0][wn * LDB + wk]), lds_off(&sW[1][wn * LDB + wk]) };

    auto stage = [&](int k0, int b) {
        async_cp16(lH[b],      gH + k0);
        async_cp16(lH[b] + 16, gH + k0 + 8);
        async_cp16(lW[b],      gW + k0);
    };

    v8f c[2][2];
#pragma unroll
    for (int a = 0; a < 2; ++a)
#pragma unroll
        for (int b = 0; b < 2; ++b)
#pragma unroll
            for (int i = 0; i < 8; ++i) c[a][b][i] = 0.0f;

    stage(0, 0);
    const int NK = FDIM / 32;
    for (int idx = 0; idx < NK; ++idx) {
        int b = idx & 1;
        wait_async0();
        __syncthreads();
        if (idx + 1 < NK) stage((idx + 1) * 32, b ^ 1);

        const __bf16* bh = sH[b];
        const __bf16* bw = sW[b];
        v16bf a0 = load_a_frag(bh + (waveM +  0) * LDA, lane);
        v16bf a1 = load_a_frag(bh + (waveM + 16) * LDA, lane);
#pragma unroll
        for (int ni = 0; ni < 2; ++ni) {
            v16bf bb = load_b_frag(bw + (waveN + ni * 16) * LDB, lane);
            c[0][ni] = __builtin_amdgcn_wmma_f32_16x16x32_bf16(false, a0, false, bb, (short)0, c[0][ni], false, false);
            c[1][ni] = __builtin_amdgcn_wmma_f32_16x16x32_bf16(false, a1, false, bb, (short)0, c[1][ni], false, false);
        }
    }

    // ---- epilogue: out[token, d] += w * y. One writer per (token,d) per
    // launch; experts run sequentially -> race-free, deterministic.
#pragma unroll
    for (int mi = 0; mi < 2; ++mi)
#pragma unroll
        for (int ni = 0; ni < 2; ++ni)
#pragma unroll
            for (int i = 0; i < 8; ++i) {
                int m = waveM + mi * 16 + ((lane >> 4) * 8 + i);
                int row = mBase + m;
                if (row < ne) {
                    int token = tidl[e * NTOK + row];
                    float w   = wgt[e * NTOK + row];
                    int d = dBase + waveN + ni * 16 + (lane & 15);
                    float* o = &out[(size_t)token * DDIM + d];
                    *o += w * c[mi][ni][i];
                }
            }
}

// ---------------------------------------------------------------------------
extern "C" void kernel_launch(void* const* d_in, const int* in_sizes, int n_in,
                              void* d_out, int out_size, void* d_ws, size_t ws_size,
                              hipStream_t stream) {
    const float* x   = (const float*)d_in[0];
    const float* gw  = (const float*)d_in[1];
    const float* wi0 = (const float*)d_in[2];
    const float* wi1 = (const float*)d_in[3];
    const float* wo  = (const float*)d_in[4];
    float* out = (float*)d_out;

    char* ws = (char*)d_ws;
    int*    cnt  = (int*)(ws + CNT_OFF);
    int*    tidl = (int*)(ws + TID_OFF);
    float*  wgt  = (float*)(ws + WGT_OFF);
    __bf16* xb   = (__bf16*)(ws + XB_OFF);
    __bf16* hb   = (__bf16*)(ws + HB_OFF);
    __bf16* wt0  = (__bf16*)(ws + WT0_OFF);
    __bf16* wt1  = (__bf16*)(ws + WT1_OFF);
    __bf16* wot  = (__bf16*)(ws + WOT_OFF);

    const int n_out = NTOK * DDIM;
    moe_init<<<(n_out + 255) / 256, 256, 0, stream>>>(out, n_out, cnt);
    moe_cvt_x<<<(NTOK * DDIM) / 256, 256, 0, stream>>>(x, xb);

    // One-time weight convert+transpose to bf16, B-fragment-major layouts.
    for (int e = 0; e < EDIM; ++e) {
        moe_cvt_trans<<<dim3(FDIM / 32, DDIM / 32), dim3(32, 8), 0, stream>>>(
            wi0 + (size_t)e * DDIM * FDIM, wt0 + (size_t)e * FDIM * DDIM, DDIM, FDIM);
        moe_cvt_trans<<<dim3(FDIM / 32, DDIM / 32), dim3(32, 8), 0, stream>>>(
            wi1 + (size_t)e * DDIM * FDIM, wt1 + (size_t)e * FDIM * DDIM, DDIM, FDIM);
        moe_cvt_trans<<<dim3(DDIM / 32, FDIM / 32), dim3(32, 8), 0, stream>>>(
            wo + (size_t)e * FDIM * DDIM, wot + (size_t)e * DDIM * FDIM, FDIM, DDIM);
    }

    moe_gate<<<NTOK / 8, 256, 0, stream>>>(x, gw, cnt, tidl, wgt);

    // Sequential experts: per-expert weights stay hot in 192 MB L2, the single
    // h buffer is reused, and out+= is race-free & deterministic.
    for (int e = 0; e < EDIM; ++e) {
        moe_mlp1<<<dim3(FDIM / 64, NTOK / 128), 256, 0, stream>>>(
            xb, wt0, wt1, cnt, tidl, hb, e);
        moe_mlp2<<<dim3(DDIM / 64, NTOK / 128), 256, 0, stream>>>(
            hb, wot, cnt, tidl, wgt, out, e);
    }
}
